// SelfAttention_45191645888943
// MI455X (gfx1250) — matrix operand order repaired and checked by reading, not compiled, not run on hardware
//
#include <hip/hip_runtime.h>

typedef __attribute__((ext_vector_type(16))) _Float16 v16h;
typedef __attribute__((ext_vector_type(8)))  float    v8f;

#define DEV __device__ __forceinline__

// ---------------------------------------------------------------------------
// WMMA fragment helpers (layouts per CDNA5 ISA §7.12.2, wave32)
// ---------------------------------------------------------------------------

// A-fragment: 16x32 f16 from row-major [row][col], leading dim ld (in halfs).
// lane m=l&15 holds row M=m; g=l>>4 selects K-halves: VGPR v -> K = 8g + (v<4?2v : 16+2(v-4)).
DEV v16h load_A(const _Float16* __restrict__ base, int ld, int m0, int k0) {
  const int l = threadIdx.x & 31;
  const int m = l & 15, g = l >> 4;
  const _Float16* row = base + (size_t)(m0 + m) * ld + k0 + g * 8;
  v16h a;
#pragma unroll
  for (int v = 0; v < 8; ++v) {
    int k = ((v & 4) << 2) + ((v & 3) << 1);   // 0,2,4,6,16,18,20,22
    a[2 * v]     = row[k];
    a[2 * v + 1] = row[k + 1];
  }
  return a;
}

// B-fragment: 32x16 f16 where B = Bt^T and Bt is row-major [n][c] (weight-style).
// lane n=l&15 holds column N=n; g=l>>4: VGPR v -> K = 8g + (v<4?2v : 16+2(v-4)) (same as A).
DEV v16h load_B(const _Float16* __restrict__ base, int ld, int n0, int k0) {
  const int l = threadIdx.x & 31;
  const int n = l & 15, g = l >> 4;
  const _Float16* row = base + (size_t)(n0 + n) * ld + k0 + g * 8;
  v16h b;
#pragma unroll
  for (int v = 0; v < 8; ++v) {
    int k = ((v & 4) << 2) + ((v & 3) << 1);   // 0,2,4,6,16,18,20,22
    b[2 * v]     = row[k];
    b[2 * v + 1] = row[k + 1];
  }
  return b;
}

DEV v8f wmma_f16(v16h a, v16h b, v8f c) {
  return __builtin_amdgcn_wmma_f32_16x16x32_f16(false, a, false, b, (short)0, c,
                                                false, false);
}

// ---------------------------------------------------------------------------
// Phase 0: fp32 -> f16 cast
// ---------------------------------------------------------------------------
__global__ void f32_to_f16_kernel(const float* __restrict__ src,
                                  _Float16* __restrict__ dst, int n) {
  int i = blockIdx.x * blockDim.x + threadIdx.x;
  if (i < n) dst[i] = (_Float16)src[i];
}

// ---------------------------------------------------------------------------
// Phase 1: QKV projection GEMM.  y[m,n] = sum_c x[m,c] * W[n,c] + bias[n]
// x rows m = ti*8 + bi  (x is (t,b,128) row-major).  n = hi*128 + di.
// transposed==0: out (b,h,t,d);  transposed==1 (V): out (b,h,d,t).
// ---------------------------------------------------------------------------
__global__ __launch_bounds__(128) void proj_kernel(
    const _Float16* __restrict__ xh,   // 8192 x 128
    const _Float16* __restrict__ W,    // 1024 x 128
    const float* __restrict__ bias,    // 1024
    _Float16* __restrict__ out, int transposed) {
  const int wave = threadIdx.x >> 5;
  const int m0 = blockIdx.x * 16;
  const int n0 = (blockIdx.y * 4 + wave) * 16;
  v8f acc = {};
#pragma unroll
  for (int kc = 0; kc < 4; ++kc) {
    v16h a = load_A(xh, 128, m0, kc * 32);
    v16h b = load_B(W, 128, n0, kc * 32);
    acc = wmma_f16(a, b, acc);
  }
  const int l = threadIdx.x & 31;
  const int n = l & 15, g = l >> 4;
  const int col = n0 + n;
  const int hi = col >> 7, di = col & 127;
  const float bv = bias[col];
#pragma unroll
  for (int r = 0; r < 8; ++r) {
    int m = m0 + r + 8 * g;          // C-layout: lanes16-31 hold M = r+8
    int ti = m >> 3, bi = m & 7;
    float val = acc[r] + bv;
    size_t idx = transposed
                     ? (((size_t)(bi * 8 + hi) * 128 + di) * 1024 + ti)
                     : (((size_t)(bi * 8 + hi) * 1024 + ti) * 128 + di);
    out[idx] = (_Float16)val;
  }
}

// ---------------------------------------------------------------------------
// Phase 2: causal flash attention per (b,h).
// Reference: raw[i,j] = k_i . q_j (i = key pos, j = query pos), mask j>i,
// softmax over j, out[i] = sum_j att[i,j] v_j.  => flash attn with K-rows as
// the "query" role.  One WG = one (b,h) x 64 i-rows, 4 waves x 16 rows each.
// Q/V^T tiles staged with GLOBAL_LOAD_ASYNC_TO_LDS_B128 (ASYNCcnt path).
// ---------------------------------------------------------------------------
#define LDQ 136   // 272B row stride: 16B aligned, bank-conflict free
#define LDV 40    // 80B row stride
#define LDP 40

__global__ __launch_bounds__(128) void attn_kernel(
    const _Float16* __restrict__ kproj,  // (b*h, 1024, 128)
    const _Float16* __restrict__ qproj,  // (b*h, 1024, 128)
    const _Float16* __restrict__ vT,     // (b*h, 128, 1024)
    _Float16* __restrict__ att)          // (8192, 1024) row = ti*8+bi
{
  __shared__ _Float16 qs[32][LDQ];
  __shared__ _Float16 vs[128][LDV];
  __shared__ _Float16 ps[4][16][LDP];

  const int bh = blockIdx.y;
  const int bi = bh >> 3, hi = bh & 7;
  const int i0 = blockIdx.x * 64;
  const int wave = threadIdx.x >> 5;
  const int i0w = i0 + wave * 16;
  const int l = threadIdx.x & 31;
  const int nn = l & 15, g = l >> 4;

  const _Float16* kbase = kproj + (size_t)bh * 1024 * 128;
  const _Float16* qbase = qproj + (size_t)bh * 1024 * 128;
  const _Float16* vbase = vT + (size_t)bh * 128 * 1024;

  // K-rows A-fragments for this wave: resident in VGPRs for the whole j loop.
  v16h ak[4];
#pragma unroll
  for (int kc = 0; kc < 4; ++kc) ak[kc] = load_A(kbase, 128, i0w, kc * 32);

  const v8f vzero = {};
  v8f o[8];
#pragma unroll
  for (int dt = 0; dt < 8; ++dt) o[dt] = vzero;
  float mrow[8], lrow[8];
#pragma unroll
  for (int r = 0; r < 8; ++r) { mrow[r] = -1e30f; lrow[r] = 0.0f; }

  const float scale = 0.08838834764831845f;  // 1/sqrt(128)
  const int jend = i0 + 63;

  // per-thread staging addresses (row assignments invariant over j loop)
  const int qrow = threadIdx.x >> 2;       // 0..31
  const int qquart = threadIdx.x & 3;      // 0..3
  const unsigned lq = (unsigned)(uintptr_t)&qs[qrow][qquart * 32];
  const unsigned lv = (unsigned)(uintptr_t)&vs[threadIdx.x][0];

  for (int jb = 0; jb <= jend; jb += 32) {
    // ---- async DMA stage: Q rows [jb, jb+32) (64B/thread) and
    //      V^T rows (d index = tid) cols [jb, jb+32) (64B/thread).
    //      NOTE: offset: immediate applies to BOTH global and LDS address.
    {
      const _Float16* gq = qbase + (size_t)(jb + qrow) * 128 + qquart * 32;
      const _Float16* gv = vbase + (size_t)threadIdx.x * 1024 + jb;
      asm volatile(
          "global_load_async_to_lds_b128 %0, %2, off\n\t"
          "global_load_async_to_lds_b128 %0, %2, off offset:16\n\t"
          "global_load_async_to_lds_b128 %0, %2, off offset:32\n\t"
          "global_load_async_to_lds_b128 %0, %2, off offset:48\n\t"
          "global_load_async_to_lds_b128 %1, %3, off\n\t"
          "global_load_async_to_lds_b128 %1, %3, off offset:16\n\t"
          "global_load_async_to_lds_b128 %1, %3, off offset:32\n\t"
          "global_load_async_to_lds_b128 %1, %3, off offset:48"
          :: "v"(lq), "v"(lv), "v"(gq), "v"(gv)
          : "memory");
      asm volatile("s_wait_asynccnt 0x0" ::: "memory");
    }
    __syncthreads();

    if (jb <= i0w + 15) {
      // S = K_i (16x128) * Q_j^T (128x32): two 16x16 N-tiles, 4 K-chunks each.
      v8f s0 = vzero, s1 = vzero;
#pragma unroll
      for (int kc = 0; kc < 4; ++kc) {
        v16h b0 = load_B(&qs[0][0], LDQ, 0, kc * 32);
        s0 = wmma_f16(ak[kc], b0, s0);
        v16h b1 = load_B(&qs[0][0], LDQ, 16, kc * 32);
        s1 = wmma_f16(ak[kc], b1, s1);
      }
      // ---- online softmax over j (rows r+8g, this lane's column nn)
      float p0v[8], p1v[8];
#pragma unroll
      for (int r = 0; r < 8; ++r) {
        int ti = i0w + r + 8 * g;
        int c0 = jb + nn, c1 = jb + 16 + nn;
        float x0 = (c0 <= ti) ? s0[r] * scale : -1e30f;
        float x1 = (c1 <= ti) ? s1[r] * scale : -1e30f;
        float mb = fmaxf(x0, x1);
        mb = fmaxf(mb, __shfl_xor(mb, 1, 32));
        mb = fmaxf(mb, __shfl_xor(mb, 2, 32));
        mb = fmaxf(mb, __shfl_xor(mb, 4, 32));
        mb = fmaxf(mb, __shfl_xor(mb, 8, 32));
        float mnew = fmaxf(mrow[r], mb);
        float al = __expf(mrow[r] - mnew);
        float p0 = __expf(x0 - mnew);
        float p1 = __expf(x1 - mnew);
        float rs = p0 + p1;
        rs += __shfl_xor(rs, 1, 32);
        rs += __shfl_xor(rs, 2, 32);
        rs += __shfl_xor(rs, 4, 32);
        rs += __shfl_xor(rs, 8, 32);
        lrow[r] = lrow[r] * al + rs;
        mrow[r] = mnew;
#pragma unroll
        for (int dt = 0; dt < 8; ++dt) o[dt][r] *= al;
        p0v[r] = p0;
        p1v[r] = p1;
      }
      // ---- re-layout P (C-format) -> A-format via per-wave LDS slice
#pragma unroll
      for (int r = 0; r < 8; ++r) {
        ps[wave][r + 8 * g][nn]      = (_Float16)p0v[r];
        ps[wave][r + 8 * g][nn + 16] = (_Float16)p1v[r];
      }
      asm volatile("s_wait_dscnt 0x0" ::: "memory");  // order LDS store->load
      v16h ap = load_A(&ps[wave][0][0], LDP, 0, 0);
      // ---- O += P (16x32) * V (32x128): 8 d-tiles
#pragma unroll
      for (int dt = 0; dt < 8; ++dt) {
        v16h bv = load_B(&vs[0][0], LDV, dt * 16, 0);
        o[dt] = wmma_f16(ap, bv, o[dt]);
      }
    }
    __syncthreads();
  }

  // ---- epilogue: normalize and store f16 att in (8192 x 1024) GEMM layout
  float linv[8];
#pragma unroll
  for (int r = 0; r < 8; ++r) linv[r] = 1.0f / lrow[r];
#pragma unroll
  for (int dt = 0; dt < 8; ++dt) {
#pragma unroll
    for (int r = 0; r < 8; ++r) {
      int ti = i0w + r + 8 * g;
      size_t idx = (size_t)(ti * 8 + bi) * 1024 + hi * 128 + dt * 16 + nn;
      att[idx] = (_Float16)(o[dt][r] * linv[r]);
    }
  }
}

// ---------------------------------------------------------------------------
// Phase 3: output projection. out[m,n] = sum_c att[m,c]*Wo[n,c] + bo[n], f32.
// ---------------------------------------------------------------------------
__global__ __launch_bounds__(128) void outproj_kernel(
    const _Float16* __restrict__ att,  // 8192 x 1024
    const _Float16* __restrict__ Woh,  // 128 x 1024
    const float* __restrict__ bo,      // 128
    float* __restrict__ out)           // 8192 x 128  == (t,b,128)
{
  const int wave = threadIdx.x >> 5;
  const int m0 = blockIdx.x * 16;
  const int n0 = (blockIdx.y * 4 + wave) * 16;
  v8f acc = {};
#pragma unroll 4
  for (int kc = 0; kc < 32; ++kc) {
    v16h a = load_A(att, 1024, m0, kc * 32);
    v16h b = load_B(Woh, 1024, n0, kc * 32);
    acc = wmma_f16(a, b, acc);
  }
  const int l = threadIdx.x & 31;
  const int n = l & 15, g = l >> 4;
  const float bv = bo[n0 + n];
#pragma unroll
  for (int r = 0; r < 8; ++r) {
    out[(size_t)(m0 + r + 8 * g) * 128 + n0 + n] = acc[r] + bv;
  }
}

// ---------------------------------------------------------------------------
extern "C" void kernel_launch(void* const* d_in, const int* in_sizes, int n_in,
                              void* d_out, int out_size, void* d_ws,
                              size_t ws_size, hipStream_t stream) {
  const float* x  = (const float*)d_in[0];
  const float* Wk = (const float*)d_in[1];
  const float* bk = (const float*)d_in[2];
  const float* Wq = (const float*)d_in[3];
  const float* bq = (const float*)d_in[4];
  const float* Wv = (const float*)d_in[5];
  const float* bv = (const float*)d_in[6];
  const float* Wo = (const float*)d_in[7];
  const float* bo = (const float*)d_in[8];
  float* out = (float*)d_out;

  // workspace partition (halfs). Total ~70 MB.
  _Float16* p = (_Float16*)d_ws;
  _Float16* xh    = p; p += 8192 * 128;
  _Float16* kproj = p; p += 8 * 8 * 1024 * 128;
  _Float16* qproj = p; p += 8 * 8 * 1024 * 128;
  _Float16* vT    = p; p += 8 * 8 * 1024 * 128;
  _Float16* att   = p; p += 8192 * 1024;
  _Float16* Wkh   = p; p += 1024 * 128;
  _Float16* Wqh   = p; p += 1024 * 128;
  _Float16* Wvh   = p; p += 1024 * 128;
  _Float16* Woh   = p; p += 1024 * 128;

  auto cvt = [&](const float* s, _Float16* d, int n) {
    f32_to_f16_kernel<<<dim3((n + 255) / 256), dim3(256), 0, stream>>>(s, d, n);
  };
  cvt(x, xh, 8192 * 128);
  cvt(Wk, Wkh, 1024 * 128);
  cvt(Wq, Wqh, 1024 * 128);
  cvt(Wv, Wvh, 1024 * 128);
  cvt(Wo, Woh, 1024 * 128);

  proj_kernel<<<dim3(512, 16), dim3(128), 0, stream>>>(xh, Wkh, bk, kproj, 0);
  proj_kernel<<<dim3(512, 16), dim3(128), 0, stream>>>(xh, Wqh, bq, qproj, 0);
  proj_kernel<<<dim3(512, 16), dim3(128), 0, stream>>>(xh, Wvh, bv, vT, 1);

  attn_kernel<<<dim3(16, 64), dim3(128), 0, stream>>>(kproj, qproj, vT, att);

  outproj_kernel<<<dim3(512, 2), dim3(128), 0, stream>>>(att, Woh, bo, out);
}
